// Attention_29145648070993
// MI455X (gfx1250) — compile-verified, matrix operand order checked by reading, not run on hardware
//
#include <hip/hip_runtime.h>
#include <hip/hip_bf16.h>

typedef __attribute__((ext_vector_type(16))) _Float16 v16h;
typedef __attribute__((ext_vector_type(8)))  _Float16 v8h;
typedef __attribute__((ext_vector_type(8)))  float    v8f;
typedef __attribute__((ext_vector_type(4)))  float    v4f;
typedef __attribute__((ext_vector_type(4)))  int      v4i;

#define NHEADS 8
#define HD     64     // head dim
#define CH     512    // channels
#define SEQ    2048   // sequence length
#define BATCH  4
#define TOK    (BATCH*SEQ)   // 8192 tokens

// WGP-scope prefetch (locality 3 -> scope WGP: fill all cache levels).
#define PREFETCH_NEAR(p) __builtin_prefetch((p), 0, 3)

// ---------------------------------------------------------------------------
// Async global->LDS staging (CDNA5 GLOBAL_LOAD_ASYNC_TO_LDS_B128, ASYNCcnt).
// Builtin signature (from hipcc diagnostic): param0 = v4i in addrspace(1),
// param1 = v4i in addrspace(3), then (int offset, int cpol).
// Guarded: falls back to plain LDS staging if the builtins aren't declared.
// ---------------------------------------------------------------------------
#if __has_builtin(__builtin_amdgcn_global_load_async_to_lds_b128) && \
    __has_builtin(__builtin_amdgcn_s_wait_asynccnt)
#define HAVE_ASYNC_LDS 1
__device__ __forceinline__ void async_cp16(_Float16* l, const _Float16* g) {
  __builtin_amdgcn_global_load_async_to_lds_b128(
      (__attribute__((address_space(1))) v4i*)g,
      (__attribute__((address_space(3))) v4i*)l, 0, 0);
}
__device__ __forceinline__ void async_wait() {
  __builtin_amdgcn_s_wait_asynccnt(0);
}
#else
#define HAVE_ASYNC_LDS 0
__device__ __forceinline__ void async_cp16(_Float16* l, const _Float16* g) {
  *(v8h*)l = *(const v8h*)g;
}
__device__ __forceinline__ void async_wait() {}
#endif

// Stage a 64-row x 64-half tile (128B rows) from global (row stride ldg
// halves) into LDS, spread over a 128-thread block: 4 x b128 per thread.
__device__ __forceinline__ void stage_tile64(_Float16* lds,
                                             const _Float16* __restrict__ g,
                                             size_t ldg, int t) {
#pragma unroll
  for (int p = 0; p < 4; ++p) {
    int c  = p * 128 + t;          // 0..511 chunk id
    int r  = c >> 3;               // row 0..63
    int s8 = (c & 7) * 8;          // 16B-aligned column start (halves)
    async_cp16(lds + r * 64 + s8, g + (size_t)r * ldg + s8);
  }
}

// ---------------------------------------------------------------------------
// A/B fragment loader for v_wmma_f32_16x16x32_f16 (wave32).
// Per ISA 7.12.2: lane L (L<16) holds row/col L, K = {0..7,16..23} for the
// lower half-wave and {8..15,24..31} for the upper half-wave; two contiguous
// 8-half (16B) groups -> two b128 loads when K is contiguous in memory.
// Works for both global and LDS-resident tiles (ds_load_b128 x2).
// ---------------------------------------------------------------------------
__device__ __forceinline__ v16h load_frag16(const _Float16* __restrict__ base,
                                            int ldk, int lane) {
  int r  = lane & 15;
  int kb = (lane >> 4) << 3;          // 0 or 8
  const _Float16* p = base + (size_t)r * ldk + kb;
  v8h lo = *(const v8h*)(p);          // K = kb .. kb+7
  v8h hi = *(const v8h*)(p + 16);     // K = kb+16 .. kb+23
  v16h f;
#pragma unroll
  for (int t = 0; t < 8; ++t) { f[t] = lo[t]; f[8 + t] = hi[t]; }
  return f;
}

// ---------------------------------------------------------------------------
// Stage 0a: fp32 -> f16 straight convert (x)
// ---------------------------------------------------------------------------
__global__ void cvt_f32_to_f16(const float* __restrict__ src,
                               _Float16* __restrict__ dst, int n) {
  for (int i = blockIdx.x * blockDim.x + threadIdx.x; i < n;
       i += gridDim.x * blockDim.x)
    dst[i] = (_Float16)src[i];
}

// ---------------------------------------------------------------------------
// Stage 0b: fp32 [K][Ncol] -> f16 transposed [Ncol][K]  (weights)
// ---------------------------------------------------------------------------
__global__ void cvt_transpose_f16(const float* __restrict__ src,
                                  _Float16* __restrict__ dst,
                                  int K, int Ncol) {
  int total = K * Ncol;
  for (int i = blockIdx.x * blockDim.x + threadIdx.x; i < total;
       i += gridDim.x * blockDim.x) {
    int n = i / K, k = i - n * K;
    dst[i] = (_Float16)src[(size_t)k * Ncol + n];
  }
}

// ---------------------------------------------------------------------------
// Stage 1: QKV = Xh @ WqkvT, fused RoPE, scatter to [B,H,N,D] (V transposed).
// grid = (TOK/64, 1536/64), block = 128 (4 waves, each wave = 16-row strip).
// ---------------------------------------------------------------------------
__global__ void qkv_rope_kernel(const _Float16* __restrict__ xh,     // [TOK][CH]
                                const _Float16* __restrict__ wT,     // [3CH][CH]
                                _Float16* __restrict__ qo,           // [B,H,N,D]
                                _Float16* __restrict__ ko,           // [B,H,N,D]
                                _Float16* __restrict__ vTo)          // [B,H,D,N]
{
  const int lane = threadIdx.x & 31, wave = threadIdx.x >> 5;
  const int row0 = blockIdx.x * 64 + wave * 16;
  const int col0 = blockIdx.y * 64;

  v8f acc[4] = {v8f{}, v8f{}, v8f{}, v8f{}};
  for (int kk = 0; kk < CH; kk += 32) {
    if (kk + 32 < CH) {   // pull next K-slab into the WGP caches
      PREFETCH_NEAR(xh + (size_t)row0 * CH + kk + 32);
      PREFETCH_NEAR(wT + (size_t)col0 * CH + kk + 32);
    }
    v16h a = load_frag16(xh + (size_t)row0 * CH + kk, CH, lane);
#pragma unroll
    for (int j = 0; j < 4; ++j) {
      v16h b = load_frag16(wT + (size_t)(col0 + j * 16) * CH + kk, CH, lane);
      acc[j] = __builtin_amdgcn_wmma_f32_16x16x32_f16(
          false, a, false, b, (short)0, acc[j], false, false);
    }
  }

  const int which = col0 / CH;          // 0=Q 1=K 2=V
  const int h     = (col0 % CH) / HD;
  const int c     = lane & 15;
  const int hip_  = lane >> 4;

  if (which < 2) {
    // RoPE: element d pairs with d^32; both live in this lane (acc[j]/acc[j+2]).
#pragma unroll
    for (int j = 0; j < 2; ++j) {
      int d1 = j * 16 + c;                                  // 0..31
      float invf = __powf(10000.0f, -(float)d1 * (1.0f / 32.0f));
#pragma unroll
      for (int i = 0; i < 8; ++i) {
        int row = row0 + i + hip_ * 8;
        int n   = row & (SEQ - 1);
        float sn, cs;
        __sincosf((float)n * invf, &sn, &cs);
        float x1 = acc[j][i], x2 = acc[j + 2][i];
        acc[j][i]     = x1 * cs - x2 * sn;
        acc[j + 2][i] = x2 * cs + x1 * sn;
      }
    }
  }

#pragma unroll
  for (int j = 0; j < 4; ++j) {
    int d = j * 16 + c;
#pragma unroll
    for (int i = 0; i < 8; ++i) {
      int row = row0 + i + hip_ * 8;
      int b   = row >> 11;                 // row / SEQ
      int n   = row & (SEQ - 1);
      _Float16 val = (_Float16)acc[j][i];
      size_t bh = (size_t)(b * NHEADS + h);
      if (which == 0)      qo[(bh * SEQ + n) * HD + d] = val;
      else if (which == 1) ko[(bh * SEQ + n) * HD + d] = val;
      else                 vTo[(bh * HD + d) * SEQ + n] = val;
    }
  }
}

// ---------------------------------------------------------------------------
// Stage 2: S = scale * Q @ K^T -> raw logits to d_out attn region, plus
// online row-max and row-sumexp. K-tiles (shared by all 4 waves) are staged
// through LDS with double-buffered async copies (ASYNCcnt overlap).
// grid = (SEQ/64, B*H), block = 128.
// ---------------------------------------------------------------------------
__global__ void scores_kernel(const _Float16* __restrict__ q,
                              const _Float16* __restrict__ k,
                              float* __restrict__ S,        // [B*H][N][N]
                              float* __restrict__ rowM,     // [B*H*N]
                              float* __restrict__ rowL)     // [B*H*N]
{
  __shared__ alignas(16) _Float16 kt[2][64 * 64];   // 2 x 8 KiB
  const int lane = threadIdx.x & 31, wave = threadIdx.x >> 5;
  const int tid  = threadIdx.x;
  const int bh = blockIdx.y;
  const int n0 = blockIdx.x * 64 + wave * 16;
  const _Float16* qb = q + (size_t)bh * SEQ * HD;
  const _Float16* kb = k + (size_t)bh * SEQ * HD;
  float* Sb = S + (size_t)bh * SEQ * SEQ;
  const float scale = 0.125f;            // D^-0.5, D=64
  const int c = lane & 15, hip_ = lane >> 4;

  // Q strip fragments reused across all 32 key chunks.
  v16h a0 = load_frag16(qb + (size_t)n0 * HD + 0, HD, lane);
  v16h a1 = load_frag16(qb + (size_t)n0 * HD + 32, HD, lane);

  // preload K tile 0
  stage_tile64(kt[0], kb, HD, tid);
  async_wait();
  __syncthreads();

  float m8[8], l8[8];
#pragma unroll
  for (int i = 0; i < 8; ++i) { m8[i] = -3.0e38f; l8[i] = 0.0f; }

  for (int it = 0; it < SEQ / 64; ++it) {
    const int m0 = it * 64;
    if (it + 1 < SEQ / 64)   // overlap next tile copy with WMMA work
      stage_tile64(kt[(it + 1) & 1], kb + (size_t)(m0 + 64) * HD, HD, tid);
    const _Float16* ktile = kt[it & 1];

    v8f acc[4] = {v8f{}, v8f{}, v8f{}, v8f{}};
#pragma unroll
    for (int j = 0; j < 4; ++j) {
      v16h b0 = load_frag16(ktile + (j * 16) * 64 + 0, 64, lane);
      acc[j] = __builtin_amdgcn_wmma_f32_16x16x32_f16(
          false, a0, false, b0, (short)0, acc[j], false, false);
      v16h b1 = load_frag16(ktile + (j * 16) * 64 + 32, 64, lane);
      acc[j] = __builtin_amdgcn_wmma_f32_16x16x32_f16(
          false, a1, false, b1, (short)0, acc[j], false, false);
    }
    // scale + store raw logits
#pragma unroll
    for (int j = 0; j < 4; ++j)
#pragma unroll
      for (int i = 0; i < 8; ++i) {
        acc[j][i] *= scale;
        Sb[(size_t)(n0 + i + hip_ * 8) * SEQ + m0 + j * 16 + c] = acc[j][i];
      }
    // online softmax statistics (lanes 0-15 vs 16-31 hold different rows;
    // xor masks < 16 stay within each half).
#pragma unroll
    for (int i = 0; i < 8; ++i) {
      float mx = acc[0][i];
#pragma unroll
      for (int j = 1; j < 4; ++j) mx = fmaxf(mx, acc[j][i]);
      for (int msk = 1; msk < 16; msk <<= 1)
        mx = fmaxf(mx, __shfl_xor(mx, msk, 32));
      float mnew = fmaxf(m8[i], mx);
      float s = 0.0f;
#pragma unroll
      for (int j = 0; j < 4; ++j) s += __expf(acc[j][i] - mnew);
      for (int msk = 1; msk < 16; msk <<= 1) s += __shfl_xor(s, msk, 32);
      l8[i] = l8[i] * __expf(m8[i] - mnew) + s;
      m8[i] = mnew;
    }
    async_wait();        // my async copies for tile it+1 landed
    __syncthreads();     // everyone done reading tile it
  }
  if ((lane & 15) == 0) {
#pragma unroll
    for (int i = 0; i < 8; ++i) {
      int row = n0 + i + hip_ * 8;
      rowM[(size_t)bh * SEQ + row] = m8[i];
      rowL[(size_t)bh * SEQ + row] = l8[i];
    }
  }
}

// ---------------------------------------------------------------------------
// Stage 3: read raw logits, normalize (write attn back), and Out_h = P @ V.
// V tiles (shared by all 4 waves) staged through LDS with async double
// buffering; the A-fragment is built straight from the global logit reads so
// each logit is read/normalized/re-written exactly once.
// grid = (SEQ/64, B*H), block = 128.
// ---------------------------------------------------------------------------
__global__ void pv_kernel(float* __restrict__ S,            // in: logits, out: attn
                          const float* __restrict__ rowM,
                          const float* __restrict__ rowL,
                          const _Float16* __restrict__ vT,  // [B,H,D,N]
                          _Float16* __restrict__ ohh)       // [TOK][CH] f16
{
  __shared__ alignas(16) _Float16 vt[2][64 * 64];   // 2 x 8 KiB
  const int lane = threadIdx.x & 31, wave = threadIdx.x >> 5;
  const int tid  = threadIdx.x;
  const int bh = blockIdx.y;
  const int b = bh / NHEADS, h = bh % NHEADS;
  const int n0 = blockIdx.x * 64 + wave * 16;
  float* Sb = S + (size_t)bh * SEQ * SEQ;
  const _Float16* vb = vT + (size_t)bh * HD * SEQ;

  const int r   = lane & 15;
  const int kb8 = (lane >> 4) << 3;
  const int row = n0 + r;                          // A-fragment row of this lane
  const float mrow = rowM[(size_t)bh * SEQ + row];
  const float rl   = 1.0f / rowL[(size_t)bh * SEQ + row];

  // preload V tile 0 (64 d-rows x 64 m, row stride SEQ)
  stage_tile64(vt[0], vb, SEQ, tid);
  async_wait();
  __syncthreads();

  v8f acc[4] = {v8f{}, v8f{}, v8f{}, v8f{}};
  for (int it = 0; it < SEQ / 64; ++it) {
    const int m0 = it * 64;
    if (it + 1 < SEQ / 64)
      stage_tile64(vt[(it + 1) & 1], vb + m0 + 64, SEQ, tid);
    const _Float16* vtile = vt[it & 1];

#pragma unroll
    for (int half = 0; half < 2; ++half) {
      const int mm = m0 + half * 32;
      float* sp = Sb + (size_t)row * SEQ + mm + kb8;
      v4f s0 = *(v4f*)(sp);
      v4f s1 = *(v4f*)(sp + 4);
      v4f s2 = *(v4f*)(sp + 16);
      v4f s3 = *(v4f*)(sp + 20);
      v16h a;
#pragma unroll
      for (int t = 0; t < 4; ++t) {
        s0[t] = __expf(s0[t] - mrow) * rl;  a[t]      = (_Float16)s0[t];
        s1[t] = __expf(s1[t] - mrow) * rl;  a[4 + t]  = (_Float16)s1[t];
        s2[t] = __expf(s2[t] - mrow) * rl;  a[8 + t]  = (_Float16)s2[t];
        s3[t] = __expf(s3[t] - mrow) * rl;  a[12 + t] = (_Float16)s3[t];
      }
      *(v4f*)(sp)      = s0;    // write normalized attention back
      *(v4f*)(sp + 4)  = s1;
      *(v4f*)(sp + 16) = s2;
      *(v4f*)(sp + 20) = s3;
#pragma unroll
      for (int j = 0; j < 4; ++j) {
        v16h bf = load_frag16(vtile + (j * 16) * 64 + half * 32, 64, lane);
        acc[j] = __builtin_amdgcn_wmma_f32_16x16x32_f16(
            false, a, false, bf, (short)0, acc[j], false, false);
      }
    }
    async_wait();
    __syncthreads();
  }
  const int cc = lane & 15, hip_ = lane >> 4;
#pragma unroll
  for (int j = 0; j < 4; ++j)
#pragma unroll
    for (int i = 0; i < 8; ++i) {
      int n = n0 + i + hip_ * 8;
      ohh[((size_t)(b * SEQ + n)) * CH + h * HD + j * 16 + cc] =
          (_Float16)acc[j][i];
    }
}

// ---------------------------------------------------------------------------
// Stage 4: out = Oh @ Wproj + b.  grid = (TOK/64, CH/64), block = 128.
// ---------------------------------------------------------------------------
__global__ void proj_kernel(const _Float16* __restrict__ ohh,   // [TOK][CH]
                            const _Float16* __restrict__ wpT,   // [CH][CH]
                            const float* __restrict__ bias,
                            float* __restrict__ out)            // [TOK][CH]
{
  const int lane = threadIdx.x & 31, wave = threadIdx.x >> 5;
  const int row0 = blockIdx.x * 64 + wave * 16;
  const int col0 = blockIdx.y * 64;
  v8f acc[4] = {v8f{}, v8f{}, v8f{}, v8f{}};
  for (int kk = 0; kk < CH; kk += 32) {
    if (kk + 32 < CH) {
      PREFETCH_NEAR(ohh + (size_t)row0 * CH + kk + 32);
      PREFETCH_NEAR(wpT + (size_t)col0 * CH + kk + 32);
    }
    v16h a = load_frag16(ohh + (size_t)row0 * CH + kk, CH, lane);
#pragma unroll
    for (int j = 0; j < 4; ++j) {
      v16h b = load_frag16(wpT + (size_t)(col0 + j * 16) * CH + kk, CH, lane);
      acc[j] = __builtin_amdgcn_wmma_f32_16x16x32_f16(
          false, a, false, b, (short)0, acc[j], false, false);
    }
  }
  const int c = lane & 15, hip_ = lane >> 4;
#pragma unroll
  for (int j = 0; j < 4; ++j) {
    float bb = bias[col0 + j * 16 + c];
#pragma unroll
    for (int i = 0; i < 8; ++i)
      out[(size_t)(row0 + i + hip_ * 8) * CH + col0 + j * 16 + c] =
          acc[j][i] + bb;
  }
}

// ---------------------------------------------------------------------------
extern "C" void kernel_launch(void* const* d_in, const int* in_sizes, int n_in,
                              void* d_out, int out_size, void* d_ws, size_t ws_size,
                              hipStream_t stream) {
  (void)in_sizes; (void)n_in; (void)out_size; (void)ws_size;
  const float* x      = (const float*)d_in[0];   // [B,N,C]
  const float* w_qkv  = (const float*)d_in[1];   // [C,3C]
  const float* w_proj = (const float*)d_in[2];   // [C,C]
  const float* b_proj = (const float*)d_in[3];   // [C]

  float* out  = (float*)d_out;                                  // [B,N,C]
  float* attn = (float*)d_out + (size_t)TOK * CH;               // [B,H,N,N]

  // workspace carve-up (f16 operands + softmax stats), 256B aligned slabs
  char* ws = (char*)d_ws;
  size_t off = 0;
  auto alloc = [&](size_t bytes) {
    void* p = ws + off;
    off += (bytes + 255) & ~(size_t)255;
    return p;
  };
  _Float16* xh    = (_Float16*)alloc((size_t)TOK * CH * 2);       // 8 MiB
  _Float16* wqkvT = (_Float16*)alloc((size_t)3 * CH * CH * 2);    // 1.5 MiB
  _Float16* wpT   = (_Float16*)alloc((size_t)CH * CH * 2);        // 0.5 MiB
  _Float16* qh    = (_Float16*)alloc((size_t)TOK * CH * 2);       // 8 MiB
  _Float16* kh    = (_Float16*)alloc((size_t)TOK * CH * 2);       // 8 MiB
  _Float16* vTh   = (_Float16*)alloc((size_t)TOK * CH * 2);       // 8 MiB
  float*    rowM  = (float*)alloc((size_t)BATCH * NHEADS * SEQ * 4);
  float*    rowL  = (float*)alloc((size_t)BATCH * NHEADS * SEQ * 4);
  _Float16* ohh   = (_Float16*)alloc((size_t)TOK * CH * 2);       // 8 MiB

  // Stage 0: precision conversion + weight transposes
  cvt_f32_to_f16<<<2048, 256, 0, stream>>>(x, xh, TOK * CH);
  cvt_transpose_f16<<<768, 256, 0, stream>>>(w_qkv, wqkvT, CH, 3 * CH);
  cvt_transpose_f16<<<256, 256, 0, stream>>>(w_proj, wpT, CH, CH);

  // Stage 1: QKV GEMM + RoPE
  qkv_rope_kernel<<<dim3(TOK / 64, (3 * CH) / 64), 128, 0, stream>>>(
      xh, wqkvT, qh, kh, vTh);

  // Stage 2: logits + softmax stats
  scores_kernel<<<dim3(SEQ / 64, BATCH * NHEADS), 128, 0, stream>>>(
      qh, kh, attn, rowM, rowL);

  // Stage 3: normalize attn in place + P@V
  pv_kernel<<<dim3(SEQ / 64, BATCH * NHEADS), 128, 0, stream>>>(
      attn, rowM, rowL, vTh, ohh);

  // Stage 4: output projection + bias
  proj_kernel<<<dim3(TOK / 64, CH / 64), 128, 0, stream>>>(
      ohh, wpT, b_proj, out);
}
